// CosineSimilarityCodebook_10101763080202
// MI455X (gfx1250) — compile-verified, matrix operand order checked by reading, not run on hardware
//
#include <hip/hip_runtime.h>

// ---------------------------------------------------------------------------
// Cosine-similarity codebook (VQ) for MI455X / gfx1250.
// GEMM 65536x2048x64 on v_wmma_f32_16x16x32_bf16 (bf16 in, f32 accumulate).
// Codebook chunks staged via GLOBAL_LOAD_ASYNC_TO_LDS_B128 (ASYNCcnt) with a
// 2-deep LDS double buffer so DMA overlaps the WMMA sweep.
// ---------------------------------------------------------------------------

typedef __bf16 bf16_t;
typedef __attribute__((ext_vector_type(16))) __bf16 v16bf;
typedef __attribute__((ext_vector_type(8)))  __bf16 v8bf;
typedef __attribute__((ext_vector_type(8)))  float  v8f;

#define NROWS   65536
#define DDIM    64
#define KCODES  2048
#define RPB     128     // rows per block
#define CHUNK   128     // codes per LDS chunk (16 KB)
#define NCHUNK  (KCODES / CHUNK)
#define DECAY   0.8f

// Async copy 16B/lane from global to LDS (no VGPR round trip; ASYNCcnt).
__device__ __forceinline__ void cp_async16(unsigned lds_off, const void* gaddr) {
    asm volatile("global_load_async_to_lds_b128 %0, %1, off"
                 :: "v"(lds_off), "v"(gaddr) : "memory");
}
__device__ __forceinline__ void wait_async_le4() {
    asm volatile("s_wait_asynccnt 0x4" ::: "memory");
}
__device__ __forceinline__ void wait_async_0() {
    asm volatile("s_wait_asynccnt 0x0" ::: "memory");
}

// Build a v16bf WMMA fragment from a row-major [*,64] bf16 tile in LDS.
// ISA 16-bit A/B layout: lane<16 -> K in {klo..klo+7, klo+16..klo+23},
// klo = 32*kk; lanes>=16 shifted by 8.
__device__ __forceinline__ v16bf load_frag(const bf16_t* tile, int r, int kk, int hi) {
    const int klo = hi * 8 + kk * 32;
    const bf16_t* p = tile + r * DDIM + klo;
    v8bf c0 = *(const v8bf*)(p);
    v8bf c1 = *(const v8bf*)(p + 16);
    v16bf o;
#pragma unroll
    for (int i = 0; i < 8; ++i) { o[i] = c0[i]; o[i + 8] = c1[i]; }
    return o;
}

// ---------------------------------------------------------------------------
// Kernel 1: normalize codebook -> f32 copy (zero-bin fallback) + bf16 (WMMA B)
// ---------------------------------------------------------------------------
__global__ void vq_prep(const float* __restrict__ emb,
                        float* __restrict__ emb_n,
                        bf16_t* __restrict__ cb) {
    const int k    = blockIdx.x * 8 + (threadIdx.x >> 5);
    const int lane = threadIdx.x & 31;
    float2 v = ((const float2*)(emb + k * DDIM))[lane];
    float ss = v.x * v.x + v.y * v.y;
#pragma unroll
    for (int m = 1; m < 32; m <<= 1) ss += __shfl_xor(ss, m, 32);
    const float rstd = rsqrtf(ss + 1e-12f);
    float2 n; n.x = v.x * rstd; n.y = v.y * rstd;
    ((float2*)(emb_n + k * DDIM))[lane] = n;
    bf16_t* cr = cb + k * DDIM + lane * 2;
    cr[0] = (bf16_t)n.x;
    cr[1] = (bf16_t)n.y;
}

// ---------------------------------------------------------------------------
// Kernel 2: fused normalize + bf16 WMMA GEMM + argmax + gather + scatter-EMA
// 256 threads = 8 waves; each wave owns a 16-row tile (128 rows / block).
// ---------------------------------------------------------------------------
__global__ void __launch_bounds__(256)
vq_main(const float* __restrict__ x,
        const float* __restrict__ emb,        // raw embeddings (gather source)
        const bf16_t* __restrict__ cb,        // normalized bf16 codebook
        float* __restrict__ quant,            // [N*64]
        float* __restrict__ idx_out,          // [N] (stored as float)
        float* __restrict__ bins,             // [K] accum (pre-zeroed)
        float* __restrict__ esum) {           // [K*64] accum (pre-zeroed)
    __shared__ bf16_t s_xbf[RPB * DDIM];           // 16 KB normalized rows
    __shared__ bf16_t s_cb[2][CHUNK * DDIM];       // 2 x 16 KB codebook chunks
    __shared__ float  s_rstd[RPB];
    __shared__ int    s_idx[RPB];

    const int tid = threadIdx.x;

    // Issue async DMA for chunk 0 immediately (overlaps the normalize phase).
    {
        const char* g = (const char*)cb;
        const unsigned lb = (unsigned)(uintptr_t)&s_cb[0][0];  // LDS offset
#pragma unroll
        for (int j = 0; j < 4; ++j) {
            const int i = tid + j * 256;
            cp_async16(lb + i * 16, g + i * 16);
        }
    }

    const int row  = tid >> 1;                // local row 0..127
    const int half = tid & 1;                 // which 32 dims
    const int grow = blockIdx.x * RPB + row;  // global row

    // ---- Phase 1: load + l2-normalize 128 rows, store bf16 to LDS ----
    const float* xr = x + grow * DDIM + half * 32;
    float ss = 0.f;
    float4 xv[8];
#pragma unroll
    for (int j = 0; j < 8; ++j) {
        xv[j] = ((const float4*)xr)[j];
        ss += xv[j].x * xv[j].x + xv[j].y * xv[j].y +
              xv[j].z * xv[j].z + xv[j].w * xv[j].w;
    }
    ss += __shfl_xor(ss, 1, 32);              // lane pairs 2i,2i+1 share a row
    const float rstd = rsqrtf(ss + 1e-12f);
    if (!half) s_rstd[row] = rstd;
    bf16_t* xd = &s_xbf[row * DDIM + half * 32];
#pragma unroll
    for (int j = 0; j < 8; ++j) {
        xd[j * 4 + 0] = (bf16_t)(xv[j].x * rstd);
        xd[j * 4 + 1] = (bf16_t)(xv[j].y * rstd);
        xd[j * 4 + 2] = (bf16_t)(xv[j].z * rstd);
        xd[j * 4 + 3] = (bf16_t)(xv[j].w * rstd);
    }
    __syncthreads();

    // ---- Phase 2: A fragments, resident for the whole column sweep ----
    const int wv      = tid >> 5;
    const int lane    = tid & 31;
    const int m       = lane & 15;
    const int hi      = lane >> 4;
    const int rowBase = wv * 16;
    v16bf a0 = load_frag(s_xbf, rowBase + m, 0, hi);
    v16bf a1 = load_frag(s_xbf, rowBase + m, 1, hi);

    float best[8];
    int   bidx[8];
#pragma unroll
    for (int r = 0; r < 8; ++r) { best[r] = -3.0e38f; bidx[r] = 0; }

    // ---- Phase 3: sweep codes; async-DMA next chunk while WMMAs run ----
    for (int cc = 0; cc < NCHUNK; ++cc) {
        const int buf = cc & 1;
        if (cc + 1 < NCHUNK) {                 // prefetch next chunk
            const char* g = (const char*)cb + (size_t)(cc + 1) * CHUNK * DDIM * 2;
            const unsigned lb = (unsigned)(uintptr_t)&s_cb[buf ^ 1][0];
#pragma unroll
            for (int j = 0; j < 4; ++j) {
                const int i = tid + j * 256;
                cp_async16(lb + i * 16, g + i * 16);
            }
            wait_async_le4();                  // 4 newest = next chunk
        } else {
            wait_async_0();
        }
        __syncthreads();                       // chunk cc visible to all waves

        const bf16_t* cbt = s_cb[buf];
        for (int t = 0; t < CHUNK / 16; t += 2) {   // 2 tiles in flight
            v16bf b00 = load_frag(cbt, (t + 0) * 16 + m, 0, hi);
            v16bf b01 = load_frag(cbt, (t + 0) * 16 + m, 1, hi);
            v16bf b10 = load_frag(cbt, (t + 1) * 16 + m, 0, hi);
            v16bf b11 = load_frag(cbt, (t + 1) * 16 + m, 1, hi);
            v8f c0 = {0.f, 0.f, 0.f, 0.f, 0.f, 0.f, 0.f, 0.f};
            v8f c1 = {0.f, 0.f, 0.f, 0.f, 0.f, 0.f, 0.f, 0.f};
            c0 = __builtin_amdgcn_wmma_f32_16x16x32_bf16(
                     false, a0, false, b00, (short)0, c0, false, false);
            c1 = __builtin_amdgcn_wmma_f32_16x16x32_bf16(
                     false, a0, false, b10, (short)0, c1, false, false);
            c0 = __builtin_amdgcn_wmma_f32_16x16x32_bf16(
                     false, a1, false, b01, (short)0, c0, false, false);
            c1 = __builtin_amdgcn_wmma_f32_16x16x32_bf16(
                     false, a1, false, b11, (short)0, c1, false, false);
            const int col0 = cc * CHUNK + (t + 0) * 16 + m;
            const int col1 = cc * CHUNK + (t + 1) * 16 + m;
#pragma unroll
            for (int r = 0; r < 8; ++r) {      // '>' keeps first (lowest) col
                const float v0 = c0[r];
                if (v0 > best[r]) { best[r] = v0; bidx[r] = col0; }
                const float v1 = c1[r];
                if (v1 > best[r]) { best[r] = v1; bidx[r] = col1; }
            }
        }
        __syncthreads();                       // done reading buf before reuse
    }

    // ---- Phase 4: cross-lane argmax within each 16-lane half ----
#pragma unroll
    for (int r = 0; r < 8; ++r) {
        float bv = best[r];
        int   bi = bidx[r];
#pragma unroll
        for (int mk = 1; mk < 16; mk <<= 1) {  // masks 1,2,4,8 stay in half
            const float ov = __shfl_xor(bv, mk, 32);
            const int   oi = __shfl_xor(bi, mk, 32);
            if (ov > bv || (ov == bv && oi < bi)) { bv = ov; bi = oi; }
        }
        if (m == 0) s_idx[rowBase + hi * 8 + r] = bi;
    }
    __syncthreads();

    // ---- Phase 5: gather + scatter stats (2 threads per row) ----
    const int   idx = s_idx[row];
    const float rs  = s_rstd[row];
    if (!half) {
        idx_out[grow] = (float)idx;
        atomicAdd(&bins[idx], 1.0f);
    }
    const float4* er = (const float4*)(emb + idx * DDIM + half * 32);
    float4*       qo = (float4*)(quant + grow * DDIM + half * 32);
    float*        es = esum + idx * DDIM + half * 32;
#pragma unroll
    for (int j = 0; j < 8; ++j) qo[j] = er[j];
#pragma unroll
    for (int j = 0; j < 32; ++j) atomicAdd(&es[j], xr[j] * rs);
}

// ---------------------------------------------------------------------------
// Kernel 3: EMA finalize. One wave per code; 2 dims per lane.
// ---------------------------------------------------------------------------
__global__ void vq_finalize(const float* __restrict__ emb,
                            const float* __restrict__ cs,
                            const float* __restrict__ bins,
                            const float* __restrict__ esum,
                            const float* __restrict__ emb_n,
                            float* __restrict__ out_cs,
                            float* __restrict__ out_emb) {
    const int k    = blockIdx.x * 8 + (threadIdx.x >> 5);
    const int lane = threadIdx.x & 31;
    const float b  = bins[k];
    if (lane == 0) out_cs[k] = cs[k] * DECAY + b * (1.0f - DECAY);
    const float bs = (b == 0.f) ? 1.f : b;
    float2 v = ((const float2*)(esum + k * DDIM))[lane];
    v.x /= bs; v.y /= bs;
    float ssq = v.x * v.x + v.y * v.y;
#pragma unroll
    for (int m = 1; m < 32; m <<= 1) ssq += __shfl_xor(ssq, m, 32);
    const float rstd = rsqrtf(ssq + 1e-12f);
    float2 en; en.x = v.x * rstd; en.y = v.y * rstd;
    if (b == 0.f) en = ((const float2*)(emb_n + k * DDIM))[lane];
    const float2 e = ((const float2*)(emb + k * DDIM))[lane];
    float2 o;
    o.x = e.x * DECAY + en.x * (1.0f - DECAY);
    o.y = e.y * DECAY + en.y * (1.0f - DECAY);
    ((float2*)(out_emb + k * DDIM))[lane] = o;
}

// ---------------------------------------------------------------------------
extern "C" void kernel_launch(void* const* d_in, const int* in_sizes, int n_in,
                              void* d_out, int out_size, void* d_ws, size_t ws_size,
                              hipStream_t stream) {
    const float* x   = (const float*)d_in[0];   // [65536*64]
    const float* emb = (const float*)d_in[1];   // [2048*64]
    const float* cs  = (const float*)d_in[2];   // [2048]

    float* out     = (float*)d_out;
    float* quant   = out;                               // 65536*64
    float* idx_out = quant + (size_t)NROWS * DDIM;      // 65536
    float* ncs     = idx_out + NROWS;                   // 2048
    float* nemb    = ncs + KCODES;                      // 2048*64

    // Workspace layout (floats): esum | bins | emb_n | cb(bf16, aliased)
    float*  esum  = (float*)d_ws;                       // 131072
    float*  bins  = esum + (size_t)KCODES * DDIM;       // 2048
    float*  emb_n = bins + KCODES;                      // 131072
    bf16_t* cb    = (bf16_t*)(emb_n + (size_t)KCODES * DDIM); // 131072 bf16

    // Accumulators must be zero every call (atomic scatter-sum targets).
    hipMemsetAsync(d_ws, 0, (size_t)(KCODES * DDIM + KCODES) * sizeof(float), stream);

    vq_prep    <<<KCODES / 8, 256, 0, stream>>>(emb, emb_n, cb);
    vq_main    <<<NROWS / RPB, 256, 0, stream>>>(x, emb, cb, quant, idx_out, bins, esum);
    vq_finalize<<<KCODES / 8, 256, 0, stream>>>(emb, cs, bins, esum, emb_n, ncs, nemb);
}
